// PolicyCompliancePredictor_32109175505594
// MI455X (gfx1250) — matrix operand order repaired and checked by reading, not compiled, not run on hardware
//
#include <hip/hip_runtime.h>
#include <hip/hip_bf16.h>

// ---------------------------------------------------------------------------
// PolicyCompliancePredictor for MI455X (gfx1250, wave32, WMMA).
// All matmuls -> v_wmma_f32_16x16x32_f16 (f16 in, f32 accumulate).
// GEMM: 128x128 block tile, 8 waves (4x2), wave tile 32x64 = 8 WMMA / k-chunk,
// double-buffered LDS (one barrier per chunk), global prefetch of next chunk.
// LSTM input-to-hidden hoisted to one big GEMM per layer; only the 128x2048x512
// recurrent GEMM + gate kernel run per timestep (600 captured launches).
// Workspace requirement: ~500 MB.
// ---------------------------------------------------------------------------

typedef _Float16 half8 __attribute__((ext_vector_type(8)));
typedef _Float16 v16h  __attribute__((ext_vector_type(16)));
typedef float    v8f   __attribute__((ext_vector_type(8)));

#define TM   128
#define TN   128
#define TK   32
#define LDSS 40   // LDS row stride in halves (32 + 8 pad)

__device__ __forceinline__ v16h frag_from_lds(const _Float16* p, int o0, int o1) {
  half8 lo = *(const half8*)(p + o0);
  half8 hi = *(const half8*)(p + o1);
  v16h r;
#pragma unroll
  for (int j = 0; j < 8; ++j) { r[j] = lo[j]; r[8 + j] = hi[j]; }
  return r;
}

// C[M,N] = alpha * (A[M,K] @ W[N,K]^T) + bias + addin(f16) + resid(f32), opt ReLU.
// Two-level batching: z = bo*BI + bi, per-pointer outer/inner strides.
__launch_bounds__(256)
__global__ void gemm_f16_wmma(
    const _Float16* __restrict__ A, int lda, long sAo, long sAi,
    const _Float16* __restrict__ W, int ldw, long sWo, long sWi,
    const float*    __restrict__ bias, long sBo,
    const _Float16* __restrict__ addin, int ldadd, long sAddo,
    const float*    __restrict__ resid,          // ld = ldc, non-batched use only
    float*          __restrict__ outf,
    _Float16*       __restrict__ outh,
    int ldc, long sCo, long sCi,
    int M, int N, int K, int BI, float alpha, int relu)
{
  __shared__ _Float16 As[2][TM * LDSS];
  __shared__ _Float16 Ws[2][TN * LDSS];

  int z  = blockIdx.z;
  int bo = z / BI, bi = z % BI;
  A += (long)bo * sAo + (long)bi * sAi;
  W += (long)bo * sWo + (long)bi * sWi;
  if (bias)  bias  += (long)bo * sBo;
  if (addin) addin += (long)bo * sAddo;
  long cOff = (long)bo * sCo + (long)bi * sCi;

  int m0 = blockIdx.y * TM;
  int n0 = blockIdx.x * TN;

  int tid  = threadIdx.x;
  int wave = tid >> 5;
  int lane = tid & 31;
  int wm   = wave >> 1;     // 0..3  (M direction, 32 rows each)
  int wn   = wave & 1;      // 0..1  (N direction, 64 cols each)

  const int ldrow = tid >> 2;        // 0..63
  const int ldcol = (tid & 3) * 8;   // 0,8,16,24

  // stage one 128x32 tile (2 rows per thread) into an LDS buffer
  auto stageA = [&](int k0, _Float16* dst) {
#pragma unroll
    for (int rr = 0; rr < 2; ++rr) {
      int r  = ldrow + rr * 64;
      int gr = m0 + r;
      half8 v = {};
      if (gr < M) {
        const _Float16* gp = A + (long)gr * lda + k0 + ldcol;
        v = *(const half8*)gp;
        if (k0 + TK < K) __builtin_prefetch(gp + TK, 0, 3);
      }
      *(half8*)(dst + r * LDSS + ldcol) = v;
    }
  };
  auto stageW = [&](int k0, _Float16* dst) {
#pragma unroll
    for (int rr = 0; rr < 2; ++rr) {
      int r  = ldrow + rr * 64;
      int gr = n0 + r;
      half8 v = {};
      if (gr < N) {
        const _Float16* gp = W + (long)gr * ldw + k0 + ldcol;
        v = *(const half8*)gp;
        if (k0 + TK < K) __builtin_prefetch(gp + TK, 0, 3);
      }
      *(half8*)(dst + r * LDSS + ldcol) = v;
    }
  };

  v8f acc[2][4] = {};

  int nk = K / TK;
  stageA(0, &As[0][0]);
  stageW(0, &Ws[0][0]);
  __syncthreads();

  int fr  = lane & 15;
  int kbA = (lane >> 4) * 8;    // A: K = {kb..kb+7, kb+16..kb+23}
  int kbB = (lane >> 4) * 8;    // B: K = {kb..kb+7, kb+16..kb+23}

  for (int kc = 0; kc < nk; ++kc) {
    int cur = kc & 1;
    if (kc + 1 < nk) {          // prefetch next chunk into the other buffer
      stageA((kc + 1) * TK, &As[cur ^ 1][0]);
      stageW((kc + 1) * TK, &Ws[cur ^ 1][0]);
    }
    v16h af[2], wf[4];
#pragma unroll
    for (int i = 0; i < 2; ++i)
      af[i] = frag_from_lds(&As[cur][(wm * 32 + i * 16 + fr) * LDSS], kbA, kbA + 16);
#pragma unroll
    for (int j = 0; j < 4; ++j)
      wf[j] = frag_from_lds(&Ws[cur][(wn * 64 + j * 16 + fr) * LDSS], kbB, kbB + 16);
#pragma unroll
    for (int mi = 0; mi < 2; ++mi)
#pragma unroll
      for (int ni = 0; ni < 4; ++ni)
        acc[mi][ni] = __builtin_amdgcn_wmma_f32_16x16x32_f16(
            false, af[mi], false, wf[ni], (short)0, acc[mi][ni], false, false);
    __syncthreads();            // one barrier per chunk (double buffered)
  }

  // ---- epilogue (C/D layout: VGPR r -> M = r + 8*(lane>=16), N = lane&15) ----
  int nl  = lane & 15;
  int mhi = (lane >> 4) * 8;
#pragma unroll
  for (int mi = 0; mi < 2; ++mi) {
#pragma unroll
    for (int ni = 0; ni < 4; ++ni) {
      int col = n0 + wn * 64 + ni * 16 + nl;
      if (col >= N) continue;
      float bv = bias ? bias[col] : 0.0f;
#pragma unroll
      for (int r = 0; r < 8; ++r) {
        int row = m0 + wm * 32 + mi * 16 + mhi + r;
        if (row >= M) continue;
        float v = acc[mi][ni][r] * alpha + bv;
        if (addin) v += (float)addin[(long)row * ldadd + col];
        long idx = cOff + (long)row * ldc + col;
        if (resid) v += resid[idx];
        if (relu && v < 0.0f) v = 0.0f;
        if (outf) outf[idx] = v;
        if (outh) outh[idx] = (_Float16)v;
      }
    }
  }
}

// ---------------- elementwise / small kernels ----------------

__global__ void f32_to_f16_k(const float* __restrict__ in, _Float16* __restrict__ out, long n) {
  long i = (long)blockIdx.x * blockDim.x + threadIdx.x;
  if (i < n) out[i] = (_Float16)in[i];
}

__global__ void add_vec_k(const float* __restrict__ a, const float* __restrict__ b,
                          float* __restrict__ o, int n) {
  int i = blockIdx.x * blockDim.x + threadIdx.x;
  if (i < n) o[i] = a[i] + b[i];
}

// gates i,f,g,o from zbuf[2][128][2048]; update c/h; scatter h into layer output.
__global__ void lstm_gate_k(const float* __restrict__ zbuf,
                            float* __restrict__ c, _Float16* __restrict__ h,
                            _Float16* __restrict__ outbuf,  // [B*T, 1024]
                            float* __restrict__ lstm_out,   // nullable (layer 2)
                            int t_fwd, int t_bwd, int T) {
  int idx = blockIdx.x * blockDim.x + threadIdx.x;
  if (idx >= 2 * 128 * 512) return;
  int dir = idx >> 16;
  int b   = (idx >> 9) & 127;
  int n   = idx & 511;
  const float* zp = zbuf + ((long)dir * 128 + b) * 2048;
  float zi = zp[n], zf = zp[512 + n], zg = zp[1024 + n], zo = zp[1536 + n];
  float si = 1.0f / (1.0f + __expf(-zi));
  float sf = 1.0f / (1.0f + __expf(-zf));
  float so = 1.0f / (1.0f + __expf(-zo));
  float cv = c[idx];
  cv = sf * cv + si * tanhf(zg);
  float hv = so * tanhf(cv);
  c[idx] = cv;
  h[idx] = (_Float16)hv;
  int t = dir ? t_bwd : t_fwd;
  long o = ((long)b * T + t) * 1024 + dir * 512 + n;
  outbuf[o] = (_Float16)hv;
  if (lstm_out) lstm_out[o] = hv;
}

// softmax over n=100 (in place, f32) + zero-padded f16 copy (npad=128).
__global__ void softmax_rows_k(float* __restrict__ s, _Float16* __restrict__ s16,
                               int rows, int n, int npad) {
  int r = blockIdx.x * blockDim.x + threadIdx.x;
  if (r >= rows) return;
  float* p = s + (long)r * n;
  float mx = -1e30f;
  for (int i = 0; i < n; ++i) mx = fmaxf(mx, p[i]);
  float sum = 0.0f;
  for (int i = 0; i < n; ++i) { float e = __expf(p[i] - mx); p[i] = e; sum += e; }
  float inv = 1.0f / sum;
  _Float16* q = s16 + (long)r * npad;
  for (int i = 0; i < n; ++i) { float v = p[i] * inv; p[i] = v; q[i] = (_Float16)v; }
  for (int i = n; i < npad; ++i) q[i] = (_Float16)0.0f;
}

// Vt[b][h][d][tp] = V[b*T+tp][h*128+d] (tp<T else 0)
__global__ void build_vt_k(const _Float16* __restrict__ V, _Float16* __restrict__ Vt, int T) {
  long i = (long)blockIdx.x * blockDim.x + threadIdx.x;
  if (i >= (long)128 * 8 * 128 * 128) return;
  int tp = i & 127;
  int d  = (i >> 7) & 127;
  int hh = (i >> 14) & 7;
  int b  = (int)(i >> 17);
  _Float16 v = (_Float16)0.0f;
  if (tp < T) v = V[((long)b * T + tp) * 1024 + hh * 128 + d];
  Vt[i] = v;
}

__launch_bounds__(256)
__global__ void layernorm_rows_k(float* __restrict__ y, const float* __restrict__ g,
                                 const float* __restrict__ be, int D) {
  __shared__ float red[256];
  float* p = y + (long)blockIdx.x * D;
  float s = 0.0f;
  for (int i = threadIdx.x; i < D; i += 256) s += p[i];
  red[threadIdx.x] = s; __syncthreads();
  for (int st = 128; st > 0; st >>= 1) { if (threadIdx.x < st) red[threadIdx.x] += red[threadIdx.x + st]; __syncthreads(); }
  float mu = red[0] / D; __syncthreads();
  float s2 = 0.0f;
  for (int i = threadIdx.x; i < D; i += 256) { float d = p[i] - mu; s2 += d * d; }
  red[threadIdx.x] = s2; __syncthreads();
  for (int st = 128; st > 0; st >>= 1) { if (threadIdx.x < st) red[threadIdx.x] += red[threadIdx.x + st]; __syncthreads(); }
  float rstd = rsqrtf(red[0] / D + 1e-5f);
  for (int i = threadIdx.x; i < D; i += 256) p[i] = (p[i] - mu) * rstd * g[i] + be[i];
}

__global__ void pool_mean_k(const float* __restrict__ y, float* __restrict__ pooled,
                            int T, int D) {
  int i = blockIdx.x * blockDim.x + threadIdx.x;
  if (i >= 128 * D) return;
  int b = i / D, d = i % D;
  float s = 0.0f;
  for (int t = 0; t < T; ++t) s += y[((long)b * T + t) * D + d];
  pooled[i] = s / T;
}

__global__ void finalize_k(const float* __restrict__ pred /* d_out[0..255] */,
                           const float* __restrict__ conf_logit,
                           float* __restrict__ out) {
  int b = blockIdx.x * blockDim.x + threadIdx.x;
  if (b >= 128) return;
  float p0 = pred[b * 2], p1 = pred[b * 2 + 1];
  float m = fmaxf(p0, p1);
  float e0 = __expf(p0 - m), e1 = __expf(p1 - m);
  out[256 + b] = e1 / (e0 + e1);                          // violation_probability
  out[384 + b] = 1.0f / (1.0f + __expf(-conf_logit[b]));  // confidence
}

// ---------------- host side ----------------

static void gemm(hipStream_t st,
                 const _Float16* A, int lda, long sAo, long sAi,
                 const _Float16* W, int ldw, long sWo, long sWi,
                 const float* bias, long sBo,
                 const _Float16* addin, int ldadd, long sAddo,
                 const float* resid,
                 float* outf, _Float16* outh, int ldc, long sCo, long sCi,
                 int M, int N, int K, int batches, int BI, float alpha, int relu) {
  dim3 g((N + TN - 1) / TN, (M + TM - 1) / TM, batches);
  gemm_f16_wmma<<<g, 256, 0, st>>>(A, lda, sAo, sAi, W, ldw, sWo, sWi, bias, sBo,
                                   addin, ldadd, sAddo, resid, outf, outh,
                                   ldc, sCo, sCi, M, N, K, BI, alpha, relu);
}

static void conv16(hipStream_t st, const float* in, _Float16* out, long n) {
  long g = (n + 255) / 256;
  f32_to_f16_k<<<dim3((unsigned)g), 256, 0, st>>>(in, out, n);
}

extern "C" void kernel_launch(void* const* d_in, const int* in_sizes, int n_in,
                              void* d_out, int out_size, void* d_ws, size_t ws_size,
                              hipStream_t stream) {
  (void)in_sizes; (void)n_in; (void)out_size; (void)ws_size;
  const int B = 128, T = 100, BT = B * T, H = 512, D2 = 1024, G4 = 2048, NH = 8, HD = 128;
  const long ATT_OFF  = 512;                       // d_out: pred(256) viol(128) conf(128)
  const long LSTM_OFF = ATT_OFF + (long)B * NH * T * T;

  float* out = (float*)d_out;
  const float* const* in = (const float* const*)d_in;

  // bump allocator over workspace
  char* base = (char*)d_ws;
  size_t off = 0;
  auto allocB = [&](size_t bytes) -> void* {
    off = (off + 255) & ~(size_t)255;
    void* p = base + off; off += bytes; return p;
  };
  auto allocH = [&](size_t n) { return (_Float16*)allocB(n * sizeof(_Float16)); };
  auto allocF = [&](size_t n) { return (float*)allocB(n * sizeof(float)); };

  // ---- workspace layout ----
  const int feDin[4] = {256, 256, 512, 1024};
  const int feDout[4] = {256, 512, 1024, 512};
  _Float16* wfe[4];
  for (int i = 0; i < 4; ++i) wfe[i] = allocH((size_t)feDout[i] * feDin[i]);
  _Float16* x016 = allocH((size_t)BT * 256);
  _Float16* a1   = allocH((size_t)BT * 256);
  _Float16* a2   = allocH((size_t)BT * 512);
  _Float16* a3   = allocH((size_t)BT * 1024);
  _Float16* feo  = allocH((size_t)BT * 512);
  _Float16* wih[3]; _Float16* whh[3]; float* bsum[3];
  for (int L = 0; L < 3; ++L) {
    int din = (L == 0) ? H : D2;
    wih[L]  = allocH((size_t)2 * G4 * din);
    whh[L]  = allocH((size_t)2 * G4 * H);
    bsum[L] = allocF((size_t)2 * G4);
  }
  _Float16* xz    = allocH((size_t)2 * BT * G4);
  _Float16* lbuf0 = allocH((size_t)BT * D2);
  _Float16* lbuf1 = allocH((size_t)BT * D2);
  _Float16* hst   = allocH((size_t)2 * B * H);
  float*    cst   = allocF((size_t)2 * B * H);
  float*    zbuf  = allocF((size_t)2 * B * G4);
  _Float16* wq = allocH((size_t)D2 * D2); _Float16* wk = allocH((size_t)D2 * D2);
  _Float16* wv = allocH((size_t)D2 * D2); _Float16* wo = allocH((size_t)D2 * D2);
  _Float16* Qb = allocH((size_t)BT * D2); _Float16* Kb = allocH((size_t)BT * D2);
  _Float16* Vb = allocH((size_t)BT * D2);
  _Float16* attn16 = allocH((size_t)B * NH * T * 128);   // T dim padded to 128 on K axis
  _Float16* Vt     = allocH((size_t)B * NH * HD * 128);
  _Float16* ctx    = allocH((size_t)BT * D2);
  float*    ybuf   = allocF((size_t)BT * D2);
  float*    pooled = allocF((size_t)B * D2);
  _Float16* pooled16 = allocH((size_t)B * D2);
  const int pDin[4] = {1024, 512, 256, 128}, pDout[4] = {512, 256, 128, 2};
  const int cDin[3] = {1024, 512, 256},      cDout[3] = {512, 256, 1};
  _Float16* wp[4]; for (int i = 0; i < 4; ++i) wp[i] = allocH((size_t)pDout[i] * pDin[i]);
  _Float16* wc[3]; for (int i = 0; i < 3; ++i) wc[i] = allocH((size_t)cDout[i] * cDin[i]);
  _Float16* hp1 = allocH((size_t)B * 512); _Float16* hp2 = allocH((size_t)B * 256);
  _Float16* hp3 = allocH((size_t)B * 128);
  _Float16* hc1 = allocH((size_t)B * 512); _Float16* hc2 = allocH((size_t)B * 256);
  float* conf_logit = allocF((size_t)B);

  // ---- weight / input conversions ----
  conv16(stream, in[0], x016, (long)BT * 256);
  for (int i = 0; i < 4; ++i) conv16(stream, in[2 + 2 * i], wfe[i], (long)feDout[i] * feDin[i]);
  for (int L = 0; L < 3; ++L) {
    int din = (L == 0) ? H : D2;
    int b0 = 10 + 8 * L;
    conv16(stream, in[b0 + 0], wih[L],                    (long)G4 * din);  // fwd wih
    conv16(stream, in[b0 + 4], wih[L] + (size_t)G4 * din, (long)G4 * din);  // bwd wih
    conv16(stream, in[b0 + 1], whh[L],                    (long)G4 * H);    // fwd whh
    conv16(stream, in[b0 + 5], whh[L] + (size_t)G4 * H,   (long)G4 * H);    // bwd whh
    add_vec_k<<<(G4 + 255) / 256, 256, 0, stream>>>(in[b0 + 2], in[b0 + 3], bsum[L], G4);
    add_vec_k<<<(G4 + 255) / 256, 256, 0, stream>>>(in[b0 + 6], in[b0 + 7], bsum[L] + G4, G4);
  }
  conv16(stream, in[34], wq, (long)D2 * D2);
  conv16(stream, in[36], wk, (long)D2 * D2);
  conv16(stream, in[38], wv, (long)D2 * D2);
  conv16(stream, in[40], wo, (long)D2 * D2);
  for (int i = 0; i < 4; ++i) conv16(stream, in[42 + 2 * i], wp[i], (long)pDout[i] * pDin[i]);
  for (int i = 0; i < 3; ++i) conv16(stream, in[50 + 2 * i], wc[i], (long)cDout[i] * cDin[i]);

  // ---- feature extractor (4x Linear+ReLU) ----
  gemm(stream, x016, 256, 0, 0, wfe[0], 256, 0, 0, in[3], 0, nullptr, 0, 0, nullptr,
       nullptr, a1, 256, 0, 0, BT, 256, 256, 1, 1, 1.0f, 1);
  gemm(stream, a1, 256, 0, 0, wfe[1], 256, 0, 0, in[5], 0, nullptr, 0, 0, nullptr,
       nullptr, a2, 512, 0, 0, BT, 512, 256, 1, 1, 1.0f, 1);
  gemm(stream, a2, 512, 0, 0, wfe[2], 512, 0, 0, in[7], 0, nullptr, 0, 0, nullptr,
       nullptr, a3, 1024, 0, 0, BT, 1024, 512, 1, 1, 1.0f, 1);
  gemm(stream, a3, 1024, 0, 0, wfe[3], 1024, 0, 0, in[9], 0, nullptr, 0, 0, nullptr,
       nullptr, feo, 512, 0, 0, BT, 512, 1024, 1, 1, 1.0f, 1);

  // ---- 3-layer BiLSTM ----
  const _Float16* linp[3] = {feo, lbuf0, lbuf1};
  _Float16* lout[3] = {lbuf0, lbuf1, lbuf0};
  for (int L = 0; L < 3; ++L) {
    int din = (L == 0) ? H : D2;
    // xz[dir] = X @ wih^T + (bih+bhh), both dirs batched (BI=1, batches=2)
    gemm(stream, linp[L], din, 0, 0, wih[L], din, (long)G4 * din, 0,
         bsum[L], G4, nullptr, 0, 0, nullptr,
         nullptr, xz, G4, (long)BT * G4, 0, BT, G4, din, 2, 1, 1.0f, 0);
    hipMemsetAsync(hst, 0, (size_t)2 * B * H * sizeof(_Float16), stream);
    hipMemsetAsync(cst, 0, (size_t)2 * B * H * sizeof(float), stream);
    for (int s = 0; s < T; ++s) {
      int tf = s, tb = T - 1 - s;
      // z[dir] = h[dir] @ whh[dir]^T + xz[dir][t_dir]
      long sAddo = (long)BT * G4 + (long)(tb - tf) * G4;
      gemm(stream, hst, H, (long)B * H, 0, whh[L], H, (long)G4 * H, 0,
           nullptr, 0, xz + (long)tf * G4, T * G4, sAddo, nullptr,
           zbuf, nullptr, G4, (long)B * G4, 0, B, G4, H, 2, 1, 1.0f, 0);
      lstm_gate_k<<<(2 * B * H + 255) / 256, 256, 0, stream>>>(
          zbuf, cst, hst, lout[L], (L == 2) ? (out + LSTM_OFF) : nullptr, tf, tb, T);
    }
  }
  _Float16* lstm16 = lout[2];  // lbuf0, [BT x 1024] f16

  // ---- multi-head self-attention ----
  gemm(stream, lstm16, D2, 0, 0, wq, D2, 0, 0, in[35], 0, nullptr, 0, 0, nullptr,
       nullptr, Qb, D2, 0, 0, BT, D2, D2, 1, 1, 1.0f, 0);
  gemm(stream, lstm16, D2, 0, 0, wk, D2, 0, 0, in[37], 0, nullptr, 0, 0, nullptr,
       nullptr, Kb, D2, 0, 0, BT, D2, D2, 1, 1, 1.0f, 0);
  gemm(stream, lstm16, D2, 0, 0, wv, D2, 0, 0, in[39], 0, nullptr, 0, 0, nullptr,
       nullptr, Vb, D2, 0, 0, BT, D2, D2, 1, 1, 1.0f, 0);
  // scores[b,h] = Q_bh @ K_bh^T / sqrt(HD)  -> d_out attention region (f32)
  gemm(stream, Qb, D2, (long)T * D2, HD, Kb, D2, (long)T * D2, HD,
       nullptr, 0, nullptr, 0, 0, nullptr,
       out + ATT_OFF, nullptr, T, (long)NH * T * T, (long)T * T,
       T, T, HD, B * NH, NH, 0.08838834764831845f, 0);
  softmax_rows_k<<<(B * NH * T + 255) / 256, 256, 0, stream>>>(
      out + ATT_OFF, attn16, B * NH * T, T, 128);
  {
    long n = (long)B * NH * HD * 128;
    build_vt_k<<<(unsigned)((n + 255) / 256), 256, 0, stream>>>(Vb, Vt, T);
  }
  // context[b,h] = attn_bh(padded K=128) @ Vt_bh^T
  gemm(stream, attn16, 128, (long)NH * T * 128, (long)T * 128,
       Vt, 128, (long)NH * HD * 128, (long)HD * 128,
       nullptr, 0, nullptr, 0, 0, nullptr,
       nullptr, ctx, D2, (long)T * D2, HD,
       T, HD, 128, B * NH, NH, 1.0f, 0);
  // O projection + residual (lstm_out f32 in d_out)
  gemm(stream, ctx, D2, 0, 0, wo, D2, 0, 0, in[41], 0, nullptr, 0, 0,
       out + LSTM_OFF, ybuf, nullptr, D2, 0, 0, BT, D2, D2, 1, 1, 1.0f, 0);
  layernorm_rows_k<<<BT, 256, 0, stream>>>(ybuf, in[56], in[57], D2);
  pool_mean_k<<<(B * D2 + 255) / 256, 256, 0, stream>>>(ybuf, pooled, T, D2);
  conv16(stream, pooled, pooled16, (long)B * D2);

  // ---- predictor head ----
  gemm(stream, pooled16, 1024, 0, 0, wp[0], 1024, 0, 0, in[43], 0, nullptr, 0, 0, nullptr,
       nullptr, hp1, 512, 0, 0, B, 512, 1024, 1, 1, 1.0f, 1);
  gemm(stream, hp1, 512, 0, 0, wp[1], 512, 0, 0, in[45], 0, nullptr, 0, 0, nullptr,
       nullptr, hp2, 256, 0, 0, B, 256, 512, 1, 1, 1.0f, 1);
  gemm(stream, hp2, 256, 0, 0, wp[2], 256, 0, 0, in[47], 0, nullptr, 0, 0, nullptr,
       nullptr, hp3, 128, 0, 0, B, 128, 256, 1, 1, 1.0f, 1);
  gemm(stream, hp3, 128, 0, 0, wp[3], 128, 0, 0, in[49], 0, nullptr, 0, 0, nullptr,
       out, nullptr, 2, 0, 0, B, 2, 128, 1, 1, 1.0f, 0);   // predictions -> d_out[0..255]

  // ---- confidence head ----
  gemm(stream, pooled16, 1024, 0, 0, wc[0], 1024, 0, 0, in[51], 0, nullptr, 0, 0, nullptr,
       nullptr, hc1, 512, 0, 0, B, 512, 1024, 1, 1, 1.0f, 1);
  gemm(stream, hc1, 512, 0, 0, wc[1], 512, 0, 0, in[53], 0, nullptr, 0, 0, nullptr,
       nullptr, hc2, 256, 0, 0, B, 256, 512, 1, 1, 1.0f, 1);
  gemm(stream, hc2, 256, 0, 0, wc[2], 256, 0, 0, in[55], 0, nullptr, 0, 0, nullptr,
       conf_logit, nullptr, 1, 0, 0, B, 1, 256, 1, 1, 1.0f, 0);

  finalize_k<<<1, 128, 0, stream>>>(out, conf_logit, out);
}
